// CumulativeRadonFeatures_38551626449480
// MI455X (gfx1250) — matrix operand order repaired and checked by reading, not compiled, not run on hardware
//
#include <hip/hip_runtime.h>
#include <hip/hip_bf16.h>

// ---------------- problem constants (match reference) ----------------
#define B_    32
#define C_    128
#define L_    4096
#define P_    100
#define PPAD  112          // P padded to 7 * 16
#define Q_    20
#define NBIN  21           // histogram bins: j = #thresholds <= a, 0..20
#define LT    256          // l-values per block
#define XS_STRIDE 136      // bf16 elems per LDS X row (136 % 8 == 0 -> 16B aligned frags)
#define WS_STRIDE 136

typedef __bf16 v4bf  __attribute__((ext_vector_type(4)));
typedef __bf16 v8bf  __attribute__((ext_vector_type(8)));
typedef __bf16 v16bf __attribute__((ext_vector_type(16)));
typedef float  v8f   __attribute__((ext_vector_type(8)));

__device__ __forceinline__ v16bf cat8(v8bf lo, v8bf hi) {
  return __builtin_shufflevector(lo, hi, 0,1,2,3,4,5,6,7,8,9,10,11,12,13,14,15);
}

// ---------------- workspace layout (bytes) ----------------
// [0)       hist_g : u32 [B_][PPAD][NBIN] = 32*112*21*4 = 301,056
// [301056)  Wb     : bf16 [PPAD][C_]      = 112*128*2   =  28,672
// [329728)  coef   : f32  [PPAD][2] (alpha, beta)       =     896
#define WS_OFF_WB   301056
#define WS_OFF_COEF 329728

// ============== kernel 0: zero hist, W -> bf16 (padded), per-p affine coefs ==============
__global__ void crf_prep(const float* __restrict__ W,
                         const float* __restrict__ mn,
                         const float* __restrict__ mx,
                         unsigned* __restrict__ hist_g,
                         __bf16* __restrict__ Wb,
                         float* __restrict__ coef) {
  int i = blockIdx.x * blockDim.x + threadIdx.x;
  if (i < B_ * PPAD * NBIN) hist_g[i] = 0u;
  if (i < PPAD * C_) {
    int p = i >> 7;                       // C_ == 128
    Wb[i] = (p < P_) ? (__bf16)W[i] : (__bf16)0.0f;
  }
  if (i < PPAD) {
    if (i < P_) {
      float a = 21.0f / (mx[i] - mn[i]);  // thr_q = mn + (mx-mn)*(q+1)/21
      coef[2 * i + 0] = a;
      coef[2 * i + 1] = -mn[i] * a;       // j = clamp(floor(a*x + b), 0, 20)
    } else {
      coef[2 * i + 0] = 0.0f;
      coef[2 * i + 1] = 0.0f;             // pad rows: bin 0, never read
    }
  }
}

// ============== kernel 1: fused bf16-WMMA projection + quantile histogram ==============
__global__ __launch_bounds__(256)
void crf_main(const float* __restrict__ X,
              const __bf16* __restrict__ Wb,
              const float* __restrict__ coef,
              unsigned* __restrict__ hist_g) {
  extern __shared__ char smem[];
  __bf16*   Xs     = (__bf16*)smem;                       // [LT][XS_STRIDE]
  __bf16*   Ws     = Xs + LT * XS_STRIDE;                 // [PPAD][WS_STRIDE]
  float*    coef_s = (float*)(Ws + PPAD * WS_STRIDE);     // [PPAD][2]
  unsigned* hist_s = (unsigned*)(coef_s + PPAD * 2);      // [PPAD][NBIN]

  const int tid = threadIdx.x;
  const int b   = blockIdx.y;
  const int l0  = blockIdx.x * LT;

  // ---- stage X[b, 0:128, l0:l0+256]: 4 channels/thread -> one ds_store_b64 ----
  const float* Xg = X + (size_t)b * (C_ * (size_t)L_) + l0;
  #pragma unroll 2
  for (int idx = tid; idx < (C_ / 4) * LT; idx += 256) {
    int c4 = idx >> 8;                 // LT == 256
    int l  = idx & (LT - 1);
    int c  = c4 * 4;
    const float* gp = Xg + (size_t)c * L_ + l;   // 4 strided rows, coalesced per row
    v4bf v;
    v[0] = (__bf16)gp[0 * L_];
    v[1] = (__bf16)gp[1 * L_];
    v[2] = (__bf16)gp[2 * L_];
    v[3] = (__bf16)gp[3 * L_];
    *(v4bf*)(&Xs[l * XS_STRIDE + c]) = v;        // 8B aligned (136 % 4 == 0)
  }
  // ---- stage bf16 W (tiny, L2-resident), 8B moves ----
  #pragma unroll 2
  for (int idx = tid; idx < PPAD * (C_ / 4); idx += 256) {
    int p = idx >> 5;                  // C_/4 == 32
    int c = (idx & 31) * 4;
    *(v4bf*)&Ws[p * WS_STRIDE + c] = *(const v4bf*)&Wb[p * C_ + c];
  }
  for (int idx = tid; idx < PPAD * 2; idx += 256) coef_s[idx] = coef[idx];
  for (int idx = tid; idx < PPAD * NBIN; idx += 256) hist_s[idx] = 0u;
  __syncthreads();

  const int wave = tid >> 5;
  const int lane = tid & 31;
  const int m    = lane & 15;          // row (A) / column-as-p (B) index
  const int half = lane >> 4;          // K-half split per ISA VGPR layout

  // 16 l-tiles / 8 waves = 2 per wave; each wave keeps A in regs + 7 accumulators
  #pragma unroll 1
  for (int ltc = 0; ltc < 2; ++ltc) {
    const int lt = wave * 2 + ltc;

    // A frag rows: Xs[lt*16+m][kc*32 + half*8 + {0..7}], [.. +16 + {0..7}]
    const __bf16* arow = &Xs[(lt * 16 + m) * XS_STRIDE + half * 8];
    // B frag rows: Ws[pt*16+m][kc*32 + half*8 + {0..7}], [.. +16 + {0..7}]
    const __bf16* brow = &Ws[m * WS_STRIDE + half * 8];

    v8f acc[7];
    #pragma unroll
    for (int pt = 0; pt < 7; ++pt) acc[pt] = (v8f){};

    #pragma unroll 1
    for (int kc = 0; kc < 4; ++kc) {
      v16bf af = cat8(*(const v8bf*)(arow + kc * 32),
                      *(const v8bf*)(arow + kc * 32 + 16));
      #pragma unroll
      for (int pt = 0; pt < 7; ++pt) {
        const __bf16* bp = brow + pt * (16 * WS_STRIDE) + kc * 32;
        v16bf bf = cat8(*(const v8bf*)bp, *(const v8bf*)(bp + 16));
        acc[pt] = __builtin_amdgcn_wmma_f32_16x16x32_bf16(
            /*neg_a=*/false, af, /*neg_b=*/false, bf,
            /*c_mod=*/(short)0, acc[pt], /*reuse_a=*/false, /*reuse_b=*/false);
      }
    }

    // ---- quantile binning: uniform thresholds -> bin index is one FMA ----
    // C/D layout: lane holds N = lane%16 (= p), 8 rows (l values) per acc
    #pragma unroll
    for (int pt = 0; pt < 7; ++pt) {
      int p = pt * 16 + m;
      float alpha = coef_s[2 * p + 0];
      float beta  = coef_s[2 * p + 1];
      unsigned* hp = &hist_s[p * NBIN];
      #pragma unroll
      for (int r = 0; r < 8; ++r) {
        float f = fmaf(acc[pt][r], alpha, beta);
        int j = (int)floorf(f);
        j = j < 0 ? 0 : (j > 20 ? 20 : j);
        atomicAdd(&hp[j], 1u);         // ds_add_u32
      }
    }
  }

  __syncthreads();
  // ---- merge block-local histogram into global (deterministic u32 atomics) ----
  unsigned* Hg = hist_g + (size_t)b * (PPAD * NBIN);
  for (int idx = tid; idx < PPAD * NBIN; idx += 256) {
    unsigned v = hist_s[idx];
    if (v) atomicAdd(&Hg[idx], v);
  }
}

// ============== kernel 2: inclusive prefix sum over bins -> cdf, /L ==============
__global__ void crf_finalize(const unsigned* __restrict__ hist_g,
                             float* __restrict__ out) {
  int i = blockIdx.x * blockDim.x + threadIdx.x;   // one (b, p) pair per thread
  if (i >= B_ * P_) return;
  int b = i / P_;
  int p = i % P_;
  const unsigned* h = hist_g + ((size_t)b * PPAD + p) * NBIN;
  float* o = out + (size_t)b * (P_ * Q_) + p * Q_;
  unsigned run = 0;
  #pragma unroll
  for (int q = 0; q < Q_; ++q) {
    run += h[q];                                   // count(a < thr_q) = sum_{j<=q} hist[j]
    o[q] = (float)run * (1.0f / (float)L_);
  }
}

extern "C" void kernel_launch(void* const* d_in, const int* in_sizes, int n_in,
                              void* d_out, int out_size, void* d_ws, size_t ws_size,
                              hipStream_t stream) {
  const float* X  = (const float*)d_in[0];   // [B, C, L]
  const float* W  = (const float*)d_in[1];   // [P, C]
  const float* mn = (const float*)d_in[2];   // [P]
  const float* mx = (const float*)d_in[3];   // [P]
  float* out = (float*)d_out;                // [B, P*Q]

  char* ws = (char*)d_ws;
  unsigned* hist_g = (unsigned*)ws;
  __bf16*   Wb     = (__bf16*)(ws + WS_OFF_WB);
  float*    coef   = (float*)(ws + WS_OFF_COEF);

  // kernel 0: covers max(B*PPAD*NBIN, PPAD*C) = 75264 items exactly
  crf_prep<<<(B_ * PPAD * NBIN) / 256, 256, 0, stream>>>(W, mn, mx, hist_g, Wb, coef);

  // kernel 1: grid = (L/LT, B); dynamic LDS = Xs + Ws + coef + hist
  size_t lds = (size_t)LT * XS_STRIDE * 2 + (size_t)PPAD * WS_STRIDE * 2
             + (size_t)PPAD * 2 * 4 + (size_t)PPAD * NBIN * 4;   // 110,400 B
  dim3 grid(L_ / LT, B_);
  crf_main<<<grid, 256, lds, stream>>>(X, Wb, coef, hist_g);

  // kernel 2: 3200 (b,p) pairs
  crf_finalize<<<(B_ * P_ + 255) / 256, 256, 0, stream>>>(hist_g, out);
}
